// ScalarGVPConv2D_87522843558203
// MI455X (gfx1250) — compile-verified, matrix operand order checked by reading, not run on hardware
//
#include <hip/hip_runtime.h>

typedef __attribute__((ext_vector_type(4)))  float          f32x4;
typedef __attribute__((ext_vector_type(4)))  unsigned short u16x4;
typedef __attribute__((ext_vector_type(4)))  unsigned int   u32x4;
typedef __attribute__((ext_vector_type(8)))  float          v8f;
typedef __attribute__((ext_vector_type(16))) __bf16         v16bf;

#define N_NODES 50000
#define N_EDGES 640000
#define D 128
#define DE 32
#define MSG_IN 160

// ---------- helpers ----------
static __device__ __forceinline__ unsigned short f2bf(float f) {
    union { float f; unsigned u; } v; v.f = f;
    unsigned r = v.u + 0x7FFFu + ((v.u >> 16) & 1u);   // round-to-nearest-even
    return (unsigned short)(r >> 16);
}

static __device__ __forceinline__ float silu(float x) {
    return x / (1.0f + __expf(-x));
}

static __device__ __forceinline__ float wave_sum(float v) {
    #pragma unroll
    for (int off = 16; off > 0; off >>= 1) v += __shfl_xor(v, off, 32);
    return v;
}

// Load a 16x32 bf16 A-fragment (or B-fragment of Wt[N][K]) from LDS.
// Lane l: row = caller-provided, kbase = (l>>4)*8; two contiguous 16B chunks
// at K = kc+kbase and K = kc+kbase+16  (matches ISA 7.12.2 16-bit A layout).
static __device__ __forceinline__ v16bf ld_frag(const unsigned short* base, int row,
                                                int stride, int kc, int lane) {
    int kb = ((lane >> 4) & 1) << 3;
    const unsigned short* p = base + row * stride + kc + kb;
    union { v16bf bf; u32x4 q[2]; } u;
    u.q[0] = *(const u32x4*)p;
    u.q[1] = *(const u32x4*)(p + 16);
    return u.bf;
}

// ---------- workspace layout (bytes) ----------
#define AGG_OFF 0u                      // 50000*128 f32 = 25,600,000
#define W1T_OFF 25600000u               // 128*160 bf16
#define W2T_OFF 25640960u               // 128*128 bf16
#define W3T_OFF 25673728u
#define W4T_OFF 25706496u

// ---------- kernel 0: zero agg + build transposed bf16 weights ----------
__global__ __launch_bounds__(256) void prep_kernel(
    const float* __restrict__ W1, const float* __restrict__ W2,
    const float* __restrict__ W3, const float* __restrict__ W4,
    float* __restrict__ agg,
    unsigned short* __restrict__ W1t, unsigned short* __restrict__ W2t,
    unsigned short* __restrict__ W3t, unsigned short* __restrict__ W4t)
{
    int idx = blockIdx.x * 256 + threadIdx.x;
    if (idx < N_NODES * D) agg[idx] = 0.0f;
    if (idx < D * MSG_IN) {                  // W1t[n][k] = W1[k][n]
        int n = idx / MSG_IN, k = idx % MSG_IN;
        W1t[idx] = f2bf(W1[k * D + n]);
    }
    if (idx < D * D) {                       // W{2,3,4}t[n][k] = W[k][n]
        int n = idx / D, k = idx % D;
        W2t[idx] = f2bf(W2[k * D + n]);
        W3t[idx] = f2bf(W3[k * D + n]);
        W4t[idx] = f2bf(W4[k * D + n]);
    }
}

// ---------- kernel 1: fused edge MLP + scatter-add ----------
// Block = 256 threads (8 waves), 256 edges. Wave w owns edge rows [32w, 32w+32):
// two 16-row M tiles sharing each B fragment (halves LDS traffic per WMMA).
__global__ __launch_bounds__(256) void edge_kernel(
    const float* __restrict__ scalar, const float* __restrict__ efeat,
    const int* __restrict__ src, const int* __restrict__ dst,
    const float* __restrict__ b1, const float* __restrict__ b2,
    const unsigned short* __restrict__ W1t, const unsigned short* __restrict__ W2t,
    float* __restrict__ agg)
{
    extern __shared__ __align__(16) unsigned char smem[];
    unsigned short* A1  = (unsigned short*)(smem);             // [256][160] bf16
    unsigned short* sW1 = (unsigned short*)(smem + 81920);     // [128][160] bf16 (Wt)
    unsigned short* sW2 = (unsigned short*)(smem + 122880);    // [128][128] bf16 (Wt)
    unsigned short* Mt  = (unsigned short*)(smem + 155648);    // [256][128] bf16
    float*          sb1 = (float*)(smem + 221184);             // [128]
    float*          sb2 = (float*)(smem + 221696);             // [128]

    const int tid = threadIdx.x;
    const int e0  = blockIdx.x * 256;

    // cooperative: weights + biases -> LDS
    {
        u32x4* dst1 = (u32x4*)sW1; const u32x4* s1 = (const u32x4*)W1t;
        for (int i = tid; i < 2560; i += 256) dst1[i] = s1[i];
        u32x4* dst2 = (u32x4*)sW2; const u32x4* s2 = (const u32x4*)W2t;
        for (int i = tid; i < 2048; i += 256) dst2[i] = s2[i];
    }
    if (tid < 128) { sb1[tid] = b1[tid]; sb2[tid] = b2[tid]; }

    // cooperative: gather A = concat(scalar[src], efeat) -> bf16 LDS
    for (int i = tid; i < (256 * MSG_IN) / 4; i += 256) {
        int elem = i * 4;
        int row  = elem / MSG_IN;
        int col  = elem % MSG_IN;
        int e    = e0 + row;
        const float* p = (col < D) ? (scalar + src[e] * D + col)
                                   : (efeat + e * DE + (col - D));
        f32x4 v = *(const f32x4*)p;
        u16x4 h = { f2bf(v[0]), f2bf(v[1]), f2bf(v[2]), f2bf(v[3]) };
        *(u16x4*)(A1 + row * MSG_IN + col) = h;
    }
    __syncthreads();

    const int lane = tid & 31;
    const int wave = tid >> 5;
    const int rowB = wave << 5;           // 32 rows per wave
    const int nlo  = lane & 15;
    const int hi   = lane >> 4;

    v8f acc0[8], acc1[8];
    #pragma unroll
    for (int t = 0; t < 8; ++t)
        #pragma unroll
        for (int r = 0; r < 8; ++r) { acc0[t][r] = 0.0f; acc1[t][r] = 0.0f; }

    // ---- GEMM1: [32 x 160] @ [160 x 128] ----
    #pragma unroll
    for (int kc = 0; kc < MSG_IN; kc += 32) {
        v16bf a0 = ld_frag(A1, rowB + nlo,      MSG_IN, kc, lane);
        v16bf a1 = ld_frag(A1, rowB + 16 + nlo, MSG_IN, kc, lane);
        #pragma unroll
        for (int nt = 0; nt < 8; ++nt) {
            v16bf b = ld_frag(sW1, nt * 16 + nlo, MSG_IN, kc, lane);
            acc0[nt] = __builtin_amdgcn_wmma_f32_16x16x32_bf16(
                false, a0, false, b, (short)0, acc0[nt], false, false);
            acc1[nt] = __builtin_amdgcn_wmma_f32_16x16x32_bf16(
                false, a1, false, b, (short)0, acc1[nt], false, false);
        }
    }

    // epilogue 1: bias + SiLU -> bf16 Mt   (D element (m,n): vgpr m&7, lane n+16*(m>>3))
    #pragma unroll
    for (int nt = 0; nt < 8; ++nt)
        #pragma unroll
        for (int r = 0; r < 8; ++r) {
            int m0 = rowB + r + (hi << 3);
            int n  = (nt << 4) + nlo;
            float x0 = silu(acc0[nt][r] + sb1[n]);
            float x1 = silu(acc1[nt][r] + sb1[n]);
            Mt[m0 * D + n]        = f2bf(x0);
            Mt[(m0 + 16) * D + n] = f2bf(x1);
            acc0[nt][r] = 0.0f;
            acc1[nt][r] = 0.0f;
        }

    // ---- GEMM2: [32 x 128] @ [128 x 128] ----
    #pragma unroll
    for (int kc = 0; kc < D; kc += 32) {
        v16bf a0 = ld_frag(Mt, rowB + nlo,      D, kc, lane);
        v16bf a1 = ld_frag(Mt, rowB + 16 + nlo, D, kc, lane);
        #pragma unroll
        for (int nt = 0; nt < 8; ++nt) {
            v16bf b = ld_frag(sW2, nt * 16 + nlo, D, kc, lane);
            acc0[nt] = __builtin_amdgcn_wmma_f32_16x16x32_bf16(
                false, a0, false, b, (short)0, acc0[nt], false, false);
            acc1[nt] = __builtin_amdgcn_wmma_f32_16x16x32_bf16(
                false, a1, false, b, (short)0, acc1[nt], false, false);
        }
    }

    // epilogue 2: bias + SiLU -> scatter-add to agg
    int dst0[8], dst1[8];
    #pragma unroll
    for (int r = 0; r < 8; ++r) {
        dst0[r] = dst[e0 + rowB + r + (hi << 3)];
        dst1[r] = dst[e0 + rowB + 16 + r + (hi << 3)];
    }
    #pragma unroll
    for (int nt = 0; nt < 8; ++nt)
        #pragma unroll
        for (int r = 0; r < 8; ++r) {
            int n = (nt << 4) + nlo;
            float x0 = silu(acc0[nt][r] + sb2[n]);
            float x1 = silu(acc1[nt][r] + sb2[n]);
            atomicAdd(&agg[dst0[r] * D + n], x0);
            atomicAdd(&agg[dst1[r] * D + n], x1);
        }
}

// ---------- kernel 2: node update (LN -> MLP -> residual -> LN) ----------
__global__ __launch_bounds__(256) void node_kernel(
    const float* __restrict__ scalar, const float* __restrict__ agg,
    const float* __restrict__ b3, const float* __restrict__ b4,
    const float* __restrict__ g1, const float* __restrict__ be1,
    const float* __restrict__ g2, const float* __restrict__ be2,
    const unsigned short* __restrict__ W3t, const unsigned short* __restrict__ W4t,
    float* __restrict__ out)
{
    extern __shared__ __align__(16) unsigned char smem[];
    float*          X   = (float*)(smem);                     // [128][128] f32
    unsigned short* ABf = (unsigned short*)(smem + 65536);    // [128][128] bf16
    unsigned short* sW3 = (unsigned short*)(smem + 98304);
    unsigned short* sW4 = (unsigned short*)(smem + 131072);
    float*          sb3 = (float*)(smem + 163840);
    float*          sb4 = (float*)(smem + 164352);
    float*          sg1 = (float*)(smem + 164864);
    float*          sB1 = (float*)(smem + 165376);
    float*          sg2 = (float*)(smem + 165888);
    float*          sB2 = (float*)(smem + 166400);

    const int tid   = threadIdx.x;
    const int node0 = blockIdx.x * 128;

    if (tid < 128) {
        sb3[tid] = b3[tid]; sb4[tid] = b4[tid];
        sg1[tid] = g1[tid]; sB1[tid] = be1[tid];
        sg2[tid] = g2[tid]; sB2[tid] = be2[tid];
    }
    {
        u32x4* d3 = (u32x4*)sW3; const u32x4* s3 = (const u32x4*)W3t;
        for (int i = tid; i < 2048; i += 256) d3[i] = s3[i];
        u32x4* d4 = (u32x4*)sW4; const u32x4* s4 = (const u32x4*)W4t;
        for (int i = tid; i < 2048; i += 256) d4[i] = s4[i];
    }

    // x = scalar + agg/10
    for (int i = tid; i < 4096; i += 256) {
        int row  = i >> 5;
        int col  = (i & 31) << 2;
        int node = node0 + row;
        f32x4 v;
        if (node < N_NODES) {
            f32x4 a = *(const f32x4*)(scalar + node * D + col);
            f32x4 g = *(const f32x4*)(agg + node * D + col);
            v = a + g * 0.1f;
        } else {
            v[0] = v[1] = v[2] = v[3] = 0.0f;
        }
        *(f32x4*)(X + row * D + col) = v;
    }
    __syncthreads();

    const int lane = tid & 31;
    const int wave = tid >> 5;
    const int rowB = wave << 4;
    const int nlo  = lane & 15;
    const int hi   = lane >> 4;

    // LN1 -> h (f32 back into X, bf16 into ABf); each wave owns 16 rows
    for (int i = 0; i < 16; ++i) {
        int row = rowB + i;
        f32x4 x = *(f32x4*)(X + row * D + (lane << 2));
        float s = wave_sum(x[0] + x[1] + x[2] + x[3]);
        float mu = s * (1.0f / 128.0f);
        f32x4 c = x - mu;
        float q = wave_sum(c[0]*c[0] + c[1]*c[1] + c[2]*c[2] + c[3]*c[3]);
        float inv = rsqrtf(q * (1.0f / 128.0f) + 1e-5f);
        f32x4 gv = *(f32x4*)(sg1 + (lane << 2));
        f32x4 bv = *(f32x4*)(sB1 + (lane << 2));
        f32x4 h = c * inv * gv + bv;
        *(f32x4*)(X + row * D + (lane << 2)) = h;
        u16x4 hb = { f2bf(h[0]), f2bf(h[1]), f2bf(h[2]), f2bf(h[3]) };
        *(u16x4*)(ABf + row * D + (lane << 2)) = hb;
    }

    v8f acc[8];
    #pragma unroll
    for (int t = 0; t < 8; ++t)
        #pragma unroll
        for (int r = 0; r < 8; ++r) acc[t][r] = 0.0f;

    // r1 = silu(h @ W3 + b3)
    #pragma unroll
    for (int kc = 0; kc < D; kc += 32) {
        v16bf a = ld_frag(ABf, rowB + nlo, D, kc, lane);
        #pragma unroll
        for (int nt = 0; nt < 8; ++nt) {
            v16bf b = ld_frag(sW3, nt * 16 + nlo, D, kc, lane);
            acc[nt] = __builtin_amdgcn_wmma_f32_16x16x32_bf16(
                false, a, false, b, (short)0, acc[nt], false, false);
        }
    }
    #pragma unroll
    for (int nt = 0; nt < 8; ++nt)
        #pragma unroll
        for (int r = 0; r < 8; ++r) {
            int m = rowB + r + (hi << 3);
            int n = (nt << 4) + nlo;
            float x = silu(acc[nt][r] + sb3[n]);
            ABf[m * D + n] = f2bf(x);
            acc[nt][r] = 0.0f;
        }

    // r2 = silu(r1 @ W4 + b4);  X += r2 (residual)
    #pragma unroll
    for (int kc = 0; kc < D; kc += 32) {
        v16bf a = ld_frag(ABf, rowB + nlo, D, kc, lane);
        #pragma unroll
        for (int nt = 0; nt < 8; ++nt) {
            v16bf b = ld_frag(sW4, nt * 16 + nlo, D, kc, lane);
            acc[nt] = __builtin_amdgcn_wmma_f32_16x16x32_bf16(
                false, a, false, b, (short)0, acc[nt], false, false);
        }
    }
    #pragma unroll
    for (int nt = 0; nt < 8; ++nt)
        #pragma unroll
        for (int r = 0; r < 8; ++r) {
            int m = rowB + r + (hi << 3);
            int n = (nt << 4) + nlo;
            float x = silu(acc[nt][r] + sb4[n]);
            X[m * D + n] += x;
        }
    __syncthreads();

    // LN2 -> out
    for (int i = 0; i < 16; ++i) {
        int row  = rowB + i;
        int node = node0 + row;
        f32x4 x = *(f32x4*)(X + row * D + (lane << 2));
        float s = wave_sum(x[0] + x[1] + x[2] + x[3]);
        float mu = s * (1.0f / 128.0f);
        f32x4 c = x - mu;
        float q = wave_sum(c[0]*c[0] + c[1]*c[1] + c[2]*c[2] + c[3]*c[3]);
        float inv = rsqrtf(q * (1.0f / 128.0f) + 1e-5f);
        f32x4 gv = *(f32x4*)(sg2 + (lane << 2));
        f32x4 bv = *(f32x4*)(sB2 + (lane << 2));
        f32x4 y = c * inv * gv + bv;
        if (node < N_NODES)
            *(f32x4*)(out + node * D + (lane << 2)) = y;
    }
}

// ---------- launch ----------
extern "C" void kernel_launch(void* const* d_in, const int* in_sizes, int n_in,
                              void* d_out, int out_size, void* d_ws, size_t ws_size,
                              hipStream_t stream) {
    const float* scalar = (const float*)d_in[0];
    const float* efeat  = (const float*)d_in[1];
    const int*   src    = (const int*)d_in[2];
    const int*   dst    = (const int*)d_in[3];
    const float* W1 = (const float*)d_in[4];  const float* b1 = (const float*)d_in[5];
    const float* W2 = (const float*)d_in[6];  const float* b2 = (const float*)d_in[7];
    const float* W3 = (const float*)d_in[8];  const float* b3 = (const float*)d_in[9];
    const float* W4 = (const float*)d_in[10]; const float* b4 = (const float*)d_in[11];
    const float* g1 = (const float*)d_in[12]; const float* be1 = (const float*)d_in[13];
    const float* g2 = (const float*)d_in[14]; const float* be2 = (const float*)d_in[15];
    float* out = (float*)d_out;

    unsigned char* ws = (unsigned char*)d_ws;
    float*          agg = (float*)(ws + AGG_OFF);
    unsigned short* W1t = (unsigned short*)(ws + W1T_OFF);
    unsigned short* W2t = (unsigned short*)(ws + W2T_OFF);
    unsigned short* W3t = (unsigned short*)(ws + W3T_OFF);
    unsigned short* W4t = (unsigned short*)(ws + W4T_OFF);

    prep_kernel<<<(N_NODES * D) / 256, 256, 0, stream>>>(
        W1, W2, W3, W4, agg, W1t, W2t, W3t, W4t);

    edge_kernel<<<N_EDGES / 256, 256, 222208, stream>>>(
        scalar, efeat, src, dst, b1, b2, W1t, W2t, agg);

    node_kernel<<<(N_NODES + 127) / 128, 256, 166912, stream>>>(
        scalar, agg, b3, b4, g1, be1, g2, be2, W3t, W4t, out);
}
